// FixedAttentionBlock_55448027791974
// MI455X (gfx1250) — matrix operand order repaired and checked by reading, not compiled, not run on hardware
//
#include <hip/hip_runtime.h>
#include <hip/hip_bf16.h>
#include <stdint.h>

// ---------------------------------------------------------------------------
// FixedAttentionBlock on MI455X (gfx1250, wave32, WMMA bf16 16x16x32 + TDM).
// B=8, C=512, heads=8, d=64, N=1024.  Workspace: 44,040,192 bytes.
// ---------------------------------------------------------------------------

typedef __bf16 bf16;
typedef __bf16 v16bf __attribute__((ext_vector_type(16)));
typedef float  v8f   __attribute__((ext_vector_type(8)));
typedef float  f32x4 __attribute__((ext_vector_type(4)));
typedef unsigned int u32x4 __attribute__((ext_vector_type(4)));
typedef int i32x8 __attribute__((ext_vector_type(8)));
typedef int i32x4 __attribute__((ext_vector_type(4)));

union V16 { v16bf v; u32x4 u[2]; };
union H8  { bf16 h[8]; u32x4 u; };

__device__ __forceinline__ v8f zero8() {
  v8f z = {0.f, 0.f, 0.f, 0.f, 0.f, 0.f, 0.f, 0.f};
  return z;
}

__device__ __forceinline__ v8f wmma_bf16(v16bf a, v16bf b, v8f c) {
  return __builtin_amdgcn_wmma_f32_16x16x32_bf16(false, a, false, b,
                                                 (short)0, c, false, false);
}

// A-tile (16x32 16-bit): lane row = lane&15; elems 0..7 at K=hi*8,
// elems 8..15 at K=16+hi*8 (ISA 7.12.2).
__device__ __forceinline__ v16bf load_a(const bf16* rowptr, int kk, int hi) {
  V16 t;
  t.u[0] = *(const u32x4*)(rowptr + kk + hi * 8);
  t.u[1] = *(const u32x4*)(rowptr + kk + 16 + hi * 8);
  return t.v;
}

// B-tile (32x16 16-bit): lane col = lane&15; elems 0..7 at K=hi*8,
// elems 8..15 at K=16+hi*8 (same order as A).
__device__ __forceinline__ v16bf load_b(const bf16* rowptr, int kk, int hi) {
  V16 t;
  t.u[0] = *(const u32x4*)(rowptr + kk + hi * 8);
  t.u[1] = *(const u32x4*)(rowptr + kk + 16 + hi * 8);
  return t.v;
}

// ---------------------------------------------------------------------------
// Tensor Data Mover: 2D tile load Global->LDS (ISA ch.8 D# layout).
// This toolchain exposes the 6-arg builtin:
//   (u32x4 group0, i32x8 group1, i32x4 group2, i32x4 group3, i32x8, i32 cpol)
// ---------------------------------------------------------------------------
#if defined(__gfx1250__) && __has_builtin(__builtin_amdgcn_tensor_load_to_lds) && \
    __has_builtin(__builtin_amdgcn_s_wait_tensorcnt)
#define USE_TDM 1
#else
#define USE_TDM 0
#endif

#if USE_TDM
__device__ __forceinline__ void tdm_load_2d(unsigned lds_off, const void* gptr,
                                            int dim0, int dim1, int stride0) {
  unsigned long long ga = (unsigned long long)(uintptr_t)gptr;
  u32x4 g0;
  g0[0] = 1u;                                   // count=1 (valid user D#)
  g0[1] = lds_off;                              // lds_addr (bytes)
  g0[2] = (unsigned)ga;                         // global_addr[31:0]
  g0[3] = (unsigned)((ga >> 32) & 0x1ffffffull) // global_addr[56:32]
          | (2u << 30);                         // type = 2 ("image")
  i32x8 g1;
  g1[0] = 1 << 16;                              // data_size = 2 bytes
  g1[1] = dim0 << 16;                           // tensor_dim0[15:0]
  g1[2] = dim1 << 16;                           // tensor_dim1[15:0]
  g1[3] = dim0 << 16;                           // tile_dim0
  g1[4] = dim1;                                 // tile_dim1 (tile_dim2=0)
  g1[5] = stride0;                              // tensor_dim0_stride[31:0]
  g1[6] = 0;
  g1[7] = 0;
  i32x4 z4;
  z4[0] = z4[1] = z4[2] = z4[3] = 0;            // groups 2/3 unused (2D)
  i32x8 z8;
  z8[0] = z8[1] = z8[2] = z8[3] = z8[4] = z8[5] = z8[6] = z8[7] = 0;
  __builtin_amdgcn_tensor_load_to_lds(g0, g1, z4, z4, z8, 0);
}
#endif

// ---------------------------------------------------------------------------
// Kernel 1: BatchNorm (eval) + transpose -> ht[b][n][c] bf16 (LDS 32x32 tiles)
// ---------------------------------------------------------------------------
__global__ __launch_bounds__(256) void k_bn_transpose(
    const float* __restrict__ x, const float* __restrict__ gamma,
    const float* __restrict__ beta, const float* __restrict__ rmean,
    const float* __restrict__ rvar, bf16* __restrict__ ht) {
  __shared__ bf16 tile[32][33];
  int blk = blockIdx.x;
  int nt = blk & 31;
  blk >>= 5;
  int ct = blk & 15;
  int b = blk >> 4;
  int tx = threadIdx.x & 31;
  int ty = threadIdx.x >> 5;
#pragma unroll
  for (int i = 0; i < 4; ++i) {
    int cl = ty + i * 8;
    int c = ct * 32 + cl;
    int n = nt * 32 + tx;
    float inv = gamma[c] * rsqrtf(rvar[c] + 1e-5f);
    float sh = beta[c] - rmean[c] * inv;
    float v = x[((size_t)b * 512 + c) * 1024 + n] * inv + sh;
    tile[cl][tx] = (bf16)v;
  }
  __syncthreads();
#pragma unroll
  for (int i = 0; i < 4; ++i) {
    int nl = ty + i * 8;
    ht[((size_t)b * 1024 + nt * 32 + nl) * 512 + ct * 32 + tx] = tile[tx][nl];
  }
}

// ---------------------------------------------------------------------------
// Kernel 2: f32 -> bf16 weight conversion (4 x 512x512).
// ---------------------------------------------------------------------------
__global__ __launch_bounds__(256) void k_wcvt(
    const float* __restrict__ wq, const float* __restrict__ wk,
    const float* __restrict__ wv, const float* __restrict__ wp,
    bf16* __restrict__ oq, bf16* __restrict__ ok, bf16* __restrict__ ov,
    bf16* __restrict__ op) {
  int i = blockIdx.x * 256 + threadIdx.x;
  oq[i] = (bf16)wq[i];
  ok[i] = (bf16)wk[i];
  ov[i] = (bf16)wv[i];
  op[i] = (bf16)wp[i];
}

// ---------------------------------------------------------------------------
// Kernel 3: fused QKV projection.
//  Q/K: wave = 16(o) x 128(n), A=W rows, B=h rows;   store packed b128 (d on M).
//  V:   transposed roles, wave = 16(n) x 128(o);     store packed b128 (m on M).
// ---------------------------------------------------------------------------
__global__ __launch_bounds__(256) void k_qkv_gemm(
    const bf16* __restrict__ ht, const bf16* __restrict__ wq,
    const bf16* __restrict__ wk, const bf16* __restrict__ wv,
    const float* __restrict__ bq, const float* __restrict__ bk,
    const float* __restrict__ bv, bf16* __restrict__ Qt,
    bf16* __restrict__ Kt, bf16* __restrict__ Vd) {
  int wid = blockIdx.x * 8 + (threadIdx.x >> 5);
  int lane = threadIdx.x & 31;
  int lo = lane & 15, hi = lane >> 4;
  int m = wid >> 11;          // 0=Q 1=K 2=V
  int rem = wid & 2047;
  int b = rem >> 8;
  const bf16* hb = ht + (size_t)b * 1024 * 512;
  size_t bh8 = (size_t)b * 8;

  v8f acc[8];
#pragma unroll
  for (int j = 0; j < 8; ++j) acc[j] = zero8();

  if (m < 2) {
    const bf16* W = (m == 0) ? wq : wk;
    const float* bias = (m == 0) ? bq : bk;
    bf16* dst = (m == 0) ? Qt : Kt;
    int obase = ((rem >> 3) & 31) * 16, nbase = (rem & 7) * 128;
    const bf16* arow = W + (size_t)(obase + lo) * 512;
    for (int kk = 0; kk < 512; kk += 32) {
      if (kk + 64 < 512) __builtin_prefetch(arow + kk + 64, 0, 1);
      v16bf a = load_a(arow, kk, hi);
#pragma unroll
      for (int j = 0; j < 8; ++j)
        acc[j] = wmma_bf16(
            a, load_b(hb + (size_t)(nbase + j * 16 + lo) * 512, kk, hi), acc[j]);
    }
    int hd = obase >> 6;
    int dbase = (obase & 63) + 8 * hi;
#pragma unroll
    for (int j = 0; j < 8; ++j) {
      int n = nbase + j * 16 + lo;
      H8 pk;
#pragma unroll
      for (int r = 0; r < 8; ++r)
        pk.h[r] = (bf16)(acc[j][r] + bias[obase + r + 8 * hi]);
      *(u32x4*)(dst + ((bh8 + hd) * 1024 + n) * 64 + dbase) = pk.u;
    }
  } else {
    // V^T[n,o] = sum_c h[n,c] * Wv[o,c]
    int nbase = ((rem >> 2) & 63) * 16, obase = (rem & 3) * 128;
    const bf16* arow = hb + (size_t)(nbase + lo) * 512;
    for (int kk = 0; kk < 512; kk += 32) {
      if (kk + 64 < 512) __builtin_prefetch(arow + kk + 64, 0, 1);
      v16bf a = load_a(arow, kk, hi);
#pragma unroll
      for (int j = 0; j < 8; ++j)
        acc[j] = wmma_bf16(
            a, load_b(wv + (size_t)(obase + j * 16 + lo) * 512, kk, hi), acc[j]);
    }
#pragma unroll
    for (int j = 0; j < 8; ++j) {
      int o = obase + j * 16 + lo;
      float bb = bv[o];
      H8 pk;
#pragma unroll
      for (int r = 0; r < 8; ++r) pk.h[r] = (bf16)(acc[j][r] + bb);
      *(u32x4*)(Vd + ((bh8 + (o >> 6)) * 64 + (o & 63)) * 1024 + nbase + 8 * hi) =
          pk.u;
    }
  }
}

// ---------------------------------------------------------------------------
// Kernel 4: attention.  Block = 8 waves = 8 query tiles of ONE (b,h); the
// shared K-slab (32x64) and V-slab (64x32) are staged into LDS once per block
// per key-step via TDM (tensor_load_to_lds + s_wait_tensorcnt), double-
// buffered.  Softmax uses the clip-to-[-50,50] bound -> row-sum only.
// O computed transposed (O^T = V * P^T) so d lands on M -> packed stores.
// ---------------------------------------------------------------------------
__global__ __launch_bounds__(256) void k_attention(
    const bf16* __restrict__ Qt, const bf16* __restrict__ Kt,
    const bf16* __restrict__ Vd, bf16* __restrict__ AOt) {
  __shared__ bf16 Kl[2][32 * 64];     // 8 KB
  __shared__ bf16 Vl[2][64 * 32];     // 8 KB
  __shared__ bf16 plds[8][16 * 32];   // 8 KB, per-wave P bounce
  __shared__ float rsl[8][16];        // per-wave row sums

  int wav = threadIdx.x >> 5;
  int wid = blockIdx.x * 8 + wav;
  int lane = threadIdx.x & 31;
  int lo = lane & 15, hi = lane >> 4;

  int qt = wid & 63;
  int h = (wid >> 6) & 7;
  int b = wid >> 9;
  size_t bh = (size_t)(b * 8 + h);
  const bf16* q = Qt + bh * 1024 * 64;
  const bf16* k = Kt + bh * 1024 * 64;
  const bf16* v = Vd + bh * 64 * 1024;
  int qbase = qt * 16;

  const bf16* qrow = q + (size_t)(qbase + lo) * 64;
  v16bf qa0 = load_a(qrow, 0, hi);
  v16bf qa1 = load_a(qrow, 32, hi);

  v8f oacc[4];
#pragma unroll
  for (int dt = 0; dt < 4; ++dt) oacc[dt] = zero8();
  float rsum[8];
#pragma unroll
  for (int r = 0; r < 8; ++r) rsum[r] = 0.f;

  auto stage = [&](int buf, int m0) {
#if USE_TDM
    if (wav == 0) {
      tdm_load_2d((unsigned)(uintptr_t)&Kl[buf][0],
                  (const void*)(k + (size_t)m0 * 64), 64, 32, 64);
      tdm_load_2d((unsigned)(uintptr_t)&Vl[buf][0],
                  (const void*)(v + m0), 32, 64, 1024);
    }
#else
    int tid = threadIdx.x;
    {
      int row = tid >> 3, c0 = (tid & 7) * 8;
      *(u32x4*)&Kl[buf][row * 64 + c0] =
          *(const u32x4*)(k + (size_t)(m0 + row) * 64 + c0);
    }
    {
      int row = tid >> 2, c0 = (tid & 3) * 8;
      *(u32x4*)&Vl[buf][row * 32 + c0] =
          *(const u32x4*)(v + (size_t)row * 1024 + m0 + c0);
    }
#endif
  };
  auto endstage = [&]() {
#if USE_TDM
    if (wav == 0) __builtin_amdgcn_s_wait_tensorcnt(0);
#endif
    __syncthreads();
  };

  stage(0, 0);
  endstage();

  const float scale = 0.125f;   // 1/sqrt(64)
  bf16* myp = plds[wav];

  for (int i = 0; i < 32; ++i) {
    int m0 = i * 32;
    if (i + 1 < 32) stage((i + 1) & 1, m0 + 32);
    const bf16* Kb = Kl[i & 1];
    const bf16* Vb = Vl[i & 1];

    // S = Q^T K (rows n, cols m), two 16-col subtiles x two d k-steps.
    v8f s0 = zero8(), s1 = zero8();
    s0 = wmma_bf16(qa0, load_b(Kb + (size_t)lo * 64, 0, hi), s0);
    s0 = wmma_bf16(qa1, load_b(Kb + (size_t)lo * 64, 32, hi), s0);
    s1 = wmma_bf16(qa0, load_b(Kb + (size_t)(16 + lo) * 64, 0, hi), s1);
    s1 = wmma_bf16(qa1, load_b(Kb + (size_t)(16 + lo) * 64, 32, hi), s1);

    // clip + exp, row sums, deposit P into LDS as [n][m].
#pragma unroll
    for (int r = 0; r < 8; ++r) {
      float e0 = __expf(fminf(fmaxf(s0[r] * scale, -50.f), 50.f));
      float e1 = __expf(fminf(fmaxf(s1[r] * scale, -50.f), 50.f));
      rsum[r] += e0 + e1;
      myp[(r + 8 * hi) * 32 + lo] = (bf16)e0;
      myp[(r + 8 * hi) * 32 + 16 + lo] = (bf16)e1;
    }
    asm volatile("s_wait_dscnt 0" ::: "memory");

    // O^T += V * P^T : A = V rows d (from LDS), B = P^T cols n (from LDS).
    v16bf pb = load_b(myp + (size_t)lo * 32, 0, hi);
#pragma unroll
    for (int dt = 0; dt < 4; ++dt) {
      v16bf va = load_a(Vb + (size_t)(dt * 16 + lo) * 32, 0, hi);
      oacc[dt] = wmma_bf16(va, pb, oacc[dt]);
    }
    endstage();
  }

  // Row sums: reduce across the 16-lane half-groups, then redistribute to the
  // n-column lanes of the transposed O layout via LDS.
#pragma unroll
  for (int r = 0; r < 8; ++r) {
#pragma unroll
    for (int off = 1; off < 16; off <<= 1)
      rsum[r] += __shfl_xor(rsum[r], off, 32);
  }
  if (lo == 0) {
#pragma unroll
    for (int r = 0; r < 8; ++r) rsl[wav][8 * hi + r] = rsum[r];
  }
  asm volatile("s_wait_dscnt 0" ::: "memory");
  float rinv = 1.0f / rsl[wav][lo];

  // O^T C-layout: row d = dt*16 + r + 8*hi, col n = qbase + lo -> d runs over
  // consecutive VGPRs => packed 16B stores into AOt[b][n][h*64+d].
  bf16* orow = AOt + ((size_t)b * 1024 + qbase + lo) * 512 + h * 64;
#pragma unroll
  for (int dt = 0; dt < 4; ++dt) {
    H8 pk;
#pragma unroll
    for (int r = 0; r < 8; ++r) pk.h[r] = (bf16)(oacc[dt][r] * rinv);
    *(u32x4*)(orow + dt * 16 + 8 * hi) = pk.u;
  }
}

// ---------------------------------------------------------------------------
// Kernel 5: output projection + bias + residual (transposed roles so n lands
// on M -> f32x4 residual loads / stores).  out^T[n,o] = sum_c AO[n,c]Wp[o,c].
// ---------------------------------------------------------------------------
__global__ __launch_bounds__(256) void k_proj_residual(
    const bf16* __restrict__ AOt, const bf16* __restrict__ wp,
    const float* __restrict__ bp, const float* __restrict__ x,
    float* __restrict__ out) {
  int wid = blockIdx.x * 8 + (threadIdx.x >> 5);
  int lane = threadIdx.x & 31;
  int lo = lane & 15, hi = lane >> 4;
  int b = wid >> 8;
  int rem = wid & 255;
  int nbase = (rem >> 2) * 16;   // 64 n-tiles of 16
  int obase = (rem & 3) * 128;   // 4 o-tiles of 128

  v8f acc[8];
#pragma unroll
  for (int j = 0; j < 8; ++j) acc[j] = zero8();

  const bf16* arow = AOt + ((size_t)b * 1024 + nbase + lo) * 512;
  for (int kk = 0; kk < 512; kk += 32) {
    if (kk + 64 < 512) __builtin_prefetch(arow + kk + 64, 0, 1);
    v16bf a = load_a(arow, kk, hi);
#pragma unroll
    for (int j = 0; j < 8; ++j)
      acc[j] = wmma_bf16(
          a, load_b(wp + (size_t)(obase + j * 16 + lo) * 512, kk, hi), acc[j]);
  }

#pragma unroll
  for (int j = 0; j < 8; ++j) {
    int o = obase + j * 16 + lo;
    float bb = bp[o];
    size_t base = ((size_t)b * 512 + o) * 1024 + nbase + 8 * hi;
    f32x4 x0 = *(const f32x4*)(x + base);
    f32x4 x1 = *(const f32x4*)(x + base + 4);
    f32x4 o0, o1;
#pragma unroll
    for (int r = 0; r < 4; ++r) {
      o0[r] = x0[r] + acc[j][r] + bb;
      o1[r] = x1[r] + acc[j][r + 4] + bb;
    }
    *(f32x4*)(out + base) = o0;
    *(f32x4*)(out + base + 4) = o1;
  }
}

// ---------------------------------------------------------------------------
extern "C" void kernel_launch(void* const* d_in, const int* in_sizes, int n_in,
                              void* d_out, int out_size, void* d_ws,
                              size_t ws_size, hipStream_t stream) {
  (void)in_sizes; (void)n_in; (void)out_size; (void)ws_size;

  const float* x     = (const float*)d_in[0];
  const float* gamma = (const float*)d_in[1];
  const float* beta  = (const float*)d_in[2];
  const float* rmean = (const float*)d_in[3];
  const float* rvar  = (const float*)d_in[4];
  const float* wq    = (const float*)d_in[5];
  const float* bq    = (const float*)d_in[6];
  const float* wk    = (const float*)d_in[7];
  const float* bk    = (const float*)d_in[8];
  const float* wv    = (const float*)d_in[9];
  const float* bv    = (const float*)d_in[10];
  const float* wp    = (const float*)d_in[11];
  const float* bp    = (const float*)d_in[12];
  float* out = (float*)d_out;

  bf16* ws  = (bf16*)d_ws;
  bf16* ht  = ws;                       // 4,194,304 elems
  bf16* wqb = ht + 4194304;
  bf16* wkb = wqb + 262144;
  bf16* wvb = wkb + 262144;
  bf16* wpb = wvb + 262144;
  bf16* Qt  = wpb + 262144;             // 4,194,304
  bf16* Kt  = Qt + 4194304;
  bf16* Vd  = Kt + 4194304;
  bf16* AOt = Vd + 4194304;

  k_bn_transpose<<<8 * 16 * 32, 256, 0, stream>>>(x, gamma, beta, rmean, rvar, ht);
  k_wcvt<<<1024, 256, 0, stream>>>(wq, wk, wv, wp, wqb, wkb, wvb, wpb);
  k_qkv_gemm<<<768, 256, 0, stream>>>(ht, wqb, wkb, wvb, bq, bk, bv, Qt, Kt, Vd);
  k_attention<<<512, 256, 0, stream>>>(Qt, Kt, Vd, AOt);
  k_proj_residual<<<256, 256, 0, stream>>>(AOt, wpb, bp, x, out);
}
